// FepNet_46342697123845
// MI455X (gfx1250) — compile-verified, matrix operand order checked by reading, not run on hardware
//
#include <hip/hip_runtime.h>
#include <hip/hip_bf16.h>

// ---------------------------------------------------------------------------
// FepNet scan for gfx1250.
//   T=512, B=64, I=H=1024.  All matrix math via v_wmma_f32_16x16x32_bf16.
//   Sequential region reduced to: spike@W_hh (64x1024x1024 bf16 GEMM) +
//   elementwise + one grid barrier per step.  x@W_ih and prob@W_hh^T are
//   hoisted into fully parallel GEMMs outside the recurrence.
//   Spike state ping-pongs through a bf16 buffer in d_ws so per-step LDS
//   staging is a raw 32KB copy -> async global->LDS path when available.
// ---------------------------------------------------------------------------

typedef __bf16 v16bf __attribute__((ext_vector_type(16)));
typedef __bf16 v8bf  __attribute__((ext_vector_type(8)));
typedef __bf16 v4bf  __attribute__((ext_vector_type(4)));
typedef float  v8f   __attribute__((ext_vector_type(8)));
typedef int    vsi4  __attribute__((vector_size(16)));   // matches builtin proto

#define T_STEPS 512
#define BATCH   64
#define HID     1024
#define BH      (BATCH * HID)          // 65536 elements per [B,H] slab

// d_out layout (floats), concatenated in reference return order
#define OUT_SPIKES  ((size_t)0)                               // [513,64,1024]
#define OUT_PREDS   (OUT_SPIKES + (size_t)(T_STEPS + 1) * BH) // [512,64,1024]
#define OUT_PROBS   (OUT_PREDS  + (size_t)T_STEPS * BH)       // [512,64,1024]
#define OUT_PROBT   (OUT_PROBS  + (size_t)T_STEPS * BH)       // [64,1024]
#define OUT_SPIKET  (OUT_PROBT  + (size_t)BH)                 // [64,1024]

// d_ws layout
#define WS_COUNTER_OFF ((size_t)6 * 1024 * 1024)              // after 3 repacks
#define WS_SPIKEBF_OFF (WS_COUNTER_OFF + 4096)                // 2 x 128KB bf16

#define ALPHA   0.9f
#define UNIMIX  0.01f

// Cast helpers for the async-to-LDS builtin: global side wants a vector ptr in
// AS(1) (prints as "__device__"), LDS side AS(3).  Route through integers to
// sidestep generic->concrete pointer addrspace cast restrictions; generic LDS
// addresses carry the LDS byte offset in their low 32 bits.
#define GPTR(p) ((__attribute__((address_space(1))) vsi4*)(uintptr_t)(p))
#define LPTR(p) ((__attribute__((address_space(3))) vsi4*)(unsigned)(uintptr_t)(p))

__device__ inline float fast_sigmoid(float z) {
    // v_rcp_f32 instead of IEEE divide: prob is unimixed into [0.005,0.995],
    // 1-ulp rcp error is far below bf16 GEMM noise.
    return __builtin_amdgcn_rcpf(1.0f + __expf(-z));
}

// ---------------------------------------------------------------------------
// Stage 16 contiguous rows x 1024 cols of fp32 -> bf16 into LDS (coalesced).
// blockDim.x must be 256.
// ---------------------------------------------------------------------------
__device__ inline void stage16x1024_f32(const float* __restrict__ src, __bf16* lds) {
    const int tid = threadIdx.x;
#pragma unroll
    for (int i = 0; i < 16; ++i) {
        const int idx = (tid + i * 256) * 4;      // 1024 elems per iter
        float4 v = *(const float4*)(src + idx);
        v4bf b = { (__bf16)v.x, (__bf16)v.y, (__bf16)v.z, (__bf16)v.w };
        *(v4bf*)(lds + idx) = b;
    }
}

// ---------------------------------------------------------------------------
// Stage 16x1024 bf16 (32KB contiguous) into LDS.  Uses the CDNA5 async
// global->LDS data path when the toolchain exposes it (ASYNCcnt-tracked),
// otherwise a plain b128 copy through VGPRs.
// ---------------------------------------------------------------------------
__device__ inline void stage16x1024_bf16(const __bf16* __restrict__ src, __bf16* lds) {
    const int tid = threadIdx.x;
#if __has_builtin(__builtin_amdgcn_global_load_async_to_lds_b128)
#pragma unroll
    for (int i = 0; i < 8; ++i) {
        const int idx = (tid + i * 256) * 8;      // 8 bf16 = 16B per lane
        __builtin_amdgcn_global_load_async_to_lds_b128(
            GPTR(src + idx), LPTR(lds + idx), 0, 0);
    }
#if __has_builtin(__builtin_amdgcn_s_wait_asynccnt)
    __builtin_amdgcn_s_wait_asynccnt(0);
#else
    asm volatile("s_wait_asynccnt 0x0" ::: "memory");
#endif
#else
#pragma unroll
    for (int i = 0; i < 8; ++i) {
        const int idx = (tid + i * 256) * 8;
        *(v8bf*)(lds + idx) = *(const v8bf*)(src + idx);
    }
#endif
}

// ---------------------------------------------------------------------------
// One 16x16 output tile, K=1024, A from LDS (16x1024 bf16 row-major),
// B from pre-swizzled fragment buffer (512 bf16 per 32x16 fragment,
// fragments ordered k-major so the K loop streams contiguously).
// ---------------------------------------------------------------------------
__device__ inline v8f wmma_k1024(const __bf16* ldsA,
                                 const __bf16* __restrict__ Brep, v8f acc) {
    const int lane  = threadIdx.x & 31;
    const int m     = lane & 15;
    const int khalf = lane >> 4;
    const __bf16* arow = ldsA + m * HID + khalf * 8;
#pragma unroll 4
    for (int kb = 0; kb < 32; ++kb) {
        v8bf alo = *(const v8bf*)(arow + kb * 32);
        v8bf ahi = *(const v8bf*)(arow + kb * 32 + 16);
        v16bf A = __builtin_shufflevector(alo, ahi,
                    0,1,2,3,4,5,6,7,8,9,10,11,12,13,14,15);
        v16bf B = *(const v16bf*)(Brep + kb * 512 + lane * 16);
        acc = __builtin_amdgcn_wmma_f32_16x16x32_bf16(
                  false, A, false, B, (short)0, acc, false, false);
    }
    return acc;
}

// ---------------------------------------------------------------------------
// init: zero spikes[0] (fp32) + bf16 spike ping buffer 0 + barrier ticket.
// grid 256 x 256.
// ---------------------------------------------------------------------------
__global__ __launch_bounds__(256) void init_kernel(float* out, __bf16* sbf0,
                                                   unsigned* counter) {
    const unsigned idx = blockIdx.x * 256u + threadIdx.x;
    out[OUT_SPIKES + idx] = 0.0f;
    sbf0[idx] = (__bf16)0.0f;
    if (idx == 0) *counter = 0u;
}

// ---------------------------------------------------------------------------
// Repack fp32 weights into bf16 WMMA B-fragment order.
//   dest[which][frag][r]  with frag = nt*32 + kb (nt: 16-wide col tile,
//   kb: 32-deep k block), r = lane*16 + e.  Lane l holds col n = nt*16+(l&15);
//   element e maps to k = kb*32 + ((l>>4)*8 + e)  (e<8)  or  +16 (e>=8),
//   matching the 16-bit B operand layout.
// grid (2048, 3) x 512 threads.  which: 0=W_ih, 1=W_hh, 2=W_hh^T.
// ---------------------------------------------------------------------------
__global__ __launch_bounds__(512) void repack_weights(const float* __restrict__ Wih,
                                                      const float* __restrict__ Whh,
                                                      __bf16* __restrict__ dst) {
    const int frag  = blockIdx.x;        // 0..2047
    const int which = blockIdx.y;        // 0..2
    const int r     = threadIdx.x;       // 0..511
    const int l = r >> 4, e = r & 15;
    const int nt = frag >> 5, kb = frag & 31;
    const int khalf = l >> 4;
    const int kk = (e < 8) ? (khalf * 8 + e) : (16 + khalf * 8 + (e - 8));
    const int k = kb * 32 + kk;
    const int n = nt * 16 + (l & 15);
    float v;
    if (which == 0)      v = Wih[(size_t)k * HID + n];
    else if (which == 1) v = Whh[(size_t)k * HID + n];
    else                 v = Whh[(size_t)n * HID + k];   // transpose
    dst[(size_t)which * HID * HID + (size_t)frag * 512 + r] = (__bf16)v;
}

// ---------------------------------------------------------------------------
// Big parallel GEMM: D[32768,1024] = A[32768,1024] (fp32, bf16-converted on
// the fly through LDS) x Brep.  SIG!=0 applies sigmoid to the result.
// grid 16384 x 256 (8 waves, one 16x16 tile each; WG shares one A tile).
// ---------------------------------------------------------------------------
template <int SIG>
__global__ __launch_bounds__(256) void big_gemm(const float* __restrict__ A,
                                                const __bf16* __restrict__ Brep,
                                                float* __restrict__ D) {
    __shared__ __align__(16) __bf16 ldsA[16 * HID];
    const int mtile = blockIdx.x >> 3;        // 0..2047
    const int ngrp  = blockIdx.x & 7;         // 0..7
    const int wave  = threadIdx.x >> 5;       // 0..7
    const int lane  = threadIdx.x & 31;
    const int nt    = ngrp * 8 + wave;        // 0..63

    stage16x1024_f32(A + (size_t)mtile * 16 * HID, ldsA);
    __syncthreads();

    v8f acc = {0.f,0.f,0.f,0.f,0.f,0.f,0.f,0.f};
    acc = wmma_k1024(ldsA, Brep + (size_t)nt * 32 * 512, acc);

    const int n = lane & 15, mbase = (lane >> 4) * 8;
    float* d = D + ((size_t)(mtile * 16 + mbase)) * HID + nt * 16 + n;
#pragma unroll
    for (int r = 0; r < 8; ++r) {
        float v = acc[r];
        if (SIG) v = fast_sigmoid(v);
        d[(size_t)r * HID] = v;
    }
}

// ---------------------------------------------------------------------------
// Deterministic per-element uniform in [0,1): integer hash of (t,b,h).
// ---------------------------------------------------------------------------
__device__ inline float rng01(unsigned s) {
    s = s * 2654435761u + 42u;
    s ^= s >> 17; s *= 0xed5ad4bbu;
    s ^= s >> 11; s *= 0xac4c1b51u;
    s ^= s >> 15; s *= 0x31848babu;
    s ^= s >> 14;
    return (float)(s >> 8) * (1.0f / 16777216.0f);
}

// ---------------------------------------------------------------------------
// Persistent sequential scan.  32 WGs x 256 threads (co-resident).
//   WG w: mtile = w&3 (16 batch rows), its 8 waves own col tiles
//   nt = (w>>2)*8 + wave.  Each wave keeps its prob state in registers for
//   all 512 steps.  Per step: async-stage bf16 spike tile to LDS, 32 WMMAs
//   against W_hh, leaky-sigmoid + unimix + Bernoulli, store prob/spike,
//   release/acquire ticket barrier across the grid.
// ---------------------------------------------------------------------------
__global__ __launch_bounds__(256) void scan_kernel(const float* __restrict__ preX,
                                                   const __bf16* __restrict__ WhhRep,
                                                   float* out, __bf16* sbf,
                                                   unsigned* counter) {
    __shared__ __align__(16) __bf16 ldsA[16 * HID];
    const int wg    = blockIdx.x;             // 0..31
    const int mt    = wg & 3;                 // batch tile
    const int ngrp  = wg >> 2;                // 0..7
    const int wave  = threadIdx.x >> 5;
    const int lane  = threadIdx.x & 31;
    const int nt    = ngrp * 8 + wave;        // 0..63
    const int n     = lane & 15;
    const int mbase = (lane >> 4) * 8;
    const unsigned numWG = gridDim.x;

    float* spikes = out + OUT_SPIKES;
    float* probs  = out + OUT_PROBS;

    v8f prob_prev = {0.f,0.f,0.f,0.f,0.f,0.f,0.f,0.f};
    const __bf16* Bfrag = WhhRep + (size_t)nt * 32 * 512;

    for (int t = 0; t < T_STEPS; ++t) {
        // --- stage spike_prev tile (bf16 ping-pong buffer; zeros at t=0)
        stage16x1024_bf16(sbf + (size_t)(t & 1) * BH + (size_t)mt * 16 * HID,
                          ldsA);
        __syncthreads();

        // --- acc = x_t@W_ih tile (precomputed) + spike_prev@W_hh tile
        v8f acc;
        const float* px = preX + ((size_t)t * BATCH + mt * 16 + mbase) * HID
                               + nt * 16 + n;
#pragma unroll
        for (int r = 0; r < 8; ++r) acc[r] = px[(size_t)r * HID];
        acc = wmma_k1024(ldsA, Bfrag, acc);

        // --- elementwise update + sampling + stores
        const int b0 = mt * 16 + mbase;
        const int h  = nt * 16 + n;
        __bf16* sdst = sbf + (size_t)((t + 1) & 1) * BH;
#pragma unroll
        for (int r = 0; r < 8; ++r) {
            float z = ALPHA * prob_prev[r] + (1.0f - ALPHA) * acc[r];
            float p = fast_sigmoid(z);
            p = (1.0f - UNIMIX) * p + UNIMIX * 0.5f;
            const int b = b0 + r;
            const unsigned seed = ((unsigned)t << 16) | ((unsigned)b << 10)
                                | (unsigned)h;
            const float samp = (rng01(seed) < p) ? 1.0f : 0.0f;
            // straight-through: forward value samp + p - p == samp
            const float spk = (samp + p) - p;
            prob_prev[r] = p;
            const size_t el = (size_t)b * HID + h;
            probs[(size_t)t * BH + el] = p;
            spikes[(size_t)(t + 1) * BH + el] = spk;
            sdst[el] = (__bf16)spk;                 // next step's GEMM input
            if (t == T_STEPS - 1) {
                out[OUT_PROBT  + el] = p;
                out[OUT_SPIKET + el] = spk;
            }
        }

        // --- grid-wide ticket barrier (release stores, acquire loads)
        __syncthreads();   // whole WG done with stores & LDS reads
        if (threadIdx.x == 0) {
            __hip_atomic_fetch_add(counter, 1u, __ATOMIC_RELEASE,
                                   __HIP_MEMORY_SCOPE_AGENT);
            const unsigned target = (unsigned)(t + 1) * numWG;
            while (__hip_atomic_load(counter, __ATOMIC_ACQUIRE,
                                     __HIP_MEMORY_SCOPE_AGENT) < target) {
                __builtin_amdgcn_s_sleep(1);
            }
        }
        __syncthreads();
    }
}

// ---------------------------------------------------------------------------
// Launch sequence (all on `stream`, graph-capture safe, deterministic):
//   init -> repack -> preX=X@W_ih (into preds region as scratch) ->
//   scan -> preds=sigmoid(probs@W_hh^T) (overwrites scratch with output)
// ---------------------------------------------------------------------------
extern "C" void kernel_launch(void* const* d_in, const int* in_sizes, int n_in,
                              void* d_out, int out_size, void* d_ws, size_t ws_size,
                              hipStream_t stream) {
    (void)in_sizes; (void)n_in; (void)out_size; (void)ws_size;
    const float* x   = (const float*)d_in[0];  // [512,64,1024]
    const float* Wih = (const float*)d_in[1];  // [1024,1024]
    const float* Whh = (const float*)d_in[2];  // [1024,1024]
    float* out = (float*)d_out;

    __bf16* wrep = (__bf16*)d_ws;                          // 3 x 1M bf16 = 6MB
    unsigned* counter = (unsigned*)((char*)d_ws + WS_COUNTER_OFF);
    __bf16* sbf = (__bf16*)((char*)d_ws + WS_SPIKEBF_OFF); // 2 x 64K bf16

    init_kernel<<<256, 256, 0, stream>>>(out, sbf, counter);

    dim3 rgrid(2048, 3);
    repack_weights<<<rgrid, 512, 0, stream>>>(Wih, Whh, wrep);

    // preX = X @ W_ih  -> scratch in preds region (same shape as preX)
    big_gemm<0><<<16384, 256, 0, stream>>>(x, wrep, out + OUT_PREDS);

    // sequential recurrence
    scan_kernel<<<32, 256, 0, stream>>>(out + OUT_PREDS,
                                        wrep + (size_t)HID * HID,
                                        out, sbf, counter);

    // preds = sigmoid(probs @ W_hh^T)  -> real preds output
    big_gemm<1><<<16384, 256, 0, stream>>>(out + OUT_PROBS,
                                           wrep + (size_t)2 * HID * HID,
                                           out + OUT_PREDS);
}
